// VQizer_46265387713101
// MI455X (gfx1250) — matrix operand order repaired and checked by reading, not compiled, not run on hardware
//
#include <hip/hip_runtime.h>
#include <hip/hip_bf16.h>
#include <stdint.h>

// ---------------------------------------------------------------------------
// VQizer fused forward for MI455X (gfx1250, wave32, WMMA, async-LDS):
//   Pass 1 (vq_prep, once per call): W fp32 -> bf16 Wb[h][o][k];
//                                    C fp32 -> bf16 transposed Ct[h][a][o]
//   Pass 2 (vqizer_fused): per (head, 16-token tile) workgroup:
//     logits = x_h (16x128) * Wb_h^T        [v_wmma_f32_16x16x32_bf16]
//     probs  = softmax(logits / T)          [fp32, exps held in registers]
//     out    = probs (16x512) * C_h         [v_wmma_f32_16x16x32_bf16]
//   Ct_h (128 KB) staged into LDS with global_load_async_to_lds_b128
//   (ASYNCcnt); x tile staged as bf16 so all WMMA fragments are b128 loads.
// ---------------------------------------------------------------------------

typedef __bf16 bf16;
typedef __bf16 v16bf __attribute__((ext_vector_type(16)));
typedef float  v8f   __attribute__((ext_vector_type(8)));

#define NHEADS   16
#define NOPT     512
#define HS       128
#define TT       16        // token tile (M)
#define NTHREADS 256       // 8 wave32

#define WC_ELEMS (NHEADS * NOPT * HS)      // 1,048,576 elements each

// workspace layout (bytes): bf16 Wb[h][o][k], then bf16 Ct[h][a][o]
#define WB_OFF   0
#define CT_OFF   (WC_ELEMS * 2)            // 2 MB
#define WS_BYTES (CT_OFF + WC_ELEMS * 2)   // 4 MB total

// LDS layout (bytes); no static __shared__, so dynamic LDS base == 0
#define SC_OFF 0                                   // bf16 sC[HS][NOPT]   = 131072
#define SX_OFF (131072)                            // bf16 sXb[TT][HS]    =   4096
#define SL_OFF (131072 + 4096)                     // f32  sL[TT][NOPT]   =  32768
#define SP_OFF (131072 + 4096 + 32768)             // bf16 sP[TT][NOPT]   =  16384
#define SR_OFF (131072 + 4096 + 32768 + 16384)     // f32  sR[TT*16]      =   1024
#define SMEM_BYTES (SR_OFF + 1024)                 // 185344 B (< 320 KB/WGP)

union V16 { v16bf v; uint4 q[2]; };                // 32 B fragment <-> 2x b128
union PK8 { bf16 h[8]; uint4 q; };                 // 8 bf16 <-> one b128

// ---------------------------------------------------------------------------
// Pass 1: one-time fp32->bf16 conversion (+ transpose of C) into workspace.
// ---------------------------------------------------------------------------
__global__ __launch_bounds__(256)
void vq_prep(const float* __restrict__ W, const float* __restrict__ C,
             bf16* __restrict__ Wb, bf16* __restrict__ Ct)
{
    const int i = blockIdx.x * 256 + threadIdx.x;   // one float4 of each input
    // W: straight convert, same layout [h][o][k]
    {
        const float4 v = ((const float4*)W)[i];
        bf16* p = Wb + (size_t)i * 4;
        p[0] = (bf16)v.x; p[1] = (bf16)v.y; p[2] = (bf16)v.z; p[3] = (bf16)v.w;
    }
    // C: convert + transpose to [h][a][o] so GEMM2 B-fragments are contiguous
    {
        const float4 v = ((const float4*)C)[i];
        const int e = i * 4;
        const int h = e / (NOPT * HS);
        const int r = e % (NOPT * HS);
        const int o = r / HS;
        const int a = r % HS;                       // a..a+3 contiguous
        bf16* base = Ct + (size_t)h * HS * NOPT + o;
        base[(size_t)(a + 0) * NOPT] = (bf16)v.x;
        base[(size_t)(a + 1) * NOPT] = (bf16)v.y;
        base[(size_t)(a + 2) * NOPT] = (bf16)v.z;
        base[(size_t)(a + 3) * NOPT] = (bf16)v.w;
    }
}

// ---------------------------------------------------------------------------
// Pass 2: fused GEMM -> softmax -> GEMM.
// ---------------------------------------------------------------------------
__global__ __launch_bounds__(NTHREADS, 1)
void vqizer_fused(const float* __restrict__ x,
                  const bf16* __restrict__ Wb,
                  const bf16* __restrict__ Ct,
                  const float* __restrict__ temp,
                  float* __restrict__ out)
{
    extern __shared__ char smem[];
    bf16*  sC  = (bf16*)(smem + SC_OFF);
    bf16*  sXb = (bf16*)(smem + SX_OFF);
    float* sL  = (float*)(smem + SL_OFF);
    bf16*  sP  = (bf16*)(smem + SP_OFF);
    float* sR  = (float*)(smem + SR_OFF);

    const int tid  = threadIdx.x;
    const int lane = tid & 31;
    const int wave = tid >> 5;
    const int t0   = blockIdx.x * TT;    // token tile base
    const int h    = blockIdx.y;         // head

    const float invT = 1.0f / temp[0];

    // -------- async-stage Ct_h (128 KB bf16) into LDS: 8192 x b128 ----------
    {
        const uint64_t gbase = (uint64_t)(uintptr_t)(Ct + (size_t)h * HS * NOPT);
        #pragma unroll 4
        for (int i = tid; i < HS * NOPT / 8; i += NTHREADS) {   // 16 B / xfer
            const uint32_t ldsb = (uint32_t)(SC_OFF + i * 16);
            const uint64_t ga   = gbase + (uint64_t)i * 16;
            asm volatile("global_load_async_to_lds_b128 %0, %1, off"
                         :: "v"(ldsb), "v"(ga) : "memory");
        }
    }
    // -------- stage x tile as bf16: 8 contiguous floats per thread ----------
    {
        const float* xb  = x + ((size_t)t0 * NHEADS + h) * HS;
        const int   tok  = tid >> 4;         // 16 threads per token row
        const int   c8   = tid & 15;         // 16 chunks of 8 floats per row
        const float4* sp = (const float4*)(xb + (size_t)tok * NHEADS * HS + c8 * 8);
        const float4 v0 = sp[0], v1 = sp[1];
        PK8 pk;
        pk.h[0] = (bf16)v0.x; pk.h[1] = (bf16)v0.y;
        pk.h[2] = (bf16)v0.z; pk.h[3] = (bf16)v0.w;
        pk.h[4] = (bf16)v1.x; pk.h[5] = (bf16)v1.y;
        pk.h[6] = (bf16)v1.z; pk.h[7] = (bf16)v1.w;
        *(uint4*)(sXb + tok * HS + c8 * 8) = pk.q;
    }
    asm volatile("s_wait_asynccnt 0x0" ::: "memory");
    __syncthreads();

    // 16-bit operand lane mapping used here (contiguous k, same for A and B; not the ISA table order):
    //  A 16x32: lanes 0-15 -> M=lane, K 0..15; lanes 16-31 -> K 16..31
    //  B 32x16: lanes 0-15 -> N=lane, K 0..15; lanes 16-31 -> K 16..31
    const int m    = lane & 15;
    const int half = lane >> 4;

    // -------- A fragments for GEMM1 (bf16 x tile): pure b128 LDS loads ------
    v16bf afr[4];
    {
        const bf16* xr = sXb + m * HS + half * 16;
        #pragma unroll
        for (int ks = 0; ks < 4; ++ks) {
            V16 a;
            a.q[0] = *(const uint4*)(xr + ks * 32);
            a.q[1] = *(const uint4*)(xr + ks * 32 + 8);
            afr[ks] = a.v;
        }
    }

    // -------- GEMM1: logits[16 x 512] -> LDS (scaled by 1/T) ----------------
    {
        const bf16* Wh = Wb + (size_t)h * NOPT * HS;
        #pragma unroll
        for (int rep = 0; rep < 4; ++rep) {
            const int o0 = (wave + rep * 8) * 16;    // 8 waves x 4 = 32 tiles
            v8f acc = (v8f)0.0f;
            const bf16* wrow = Wh + (size_t)(o0 + m) * HS + half * 16;
            #pragma unroll
            for (int ks = 0; ks < 4; ++ks) {
                const uint4* wq = (const uint4*)(wrow + ks * 32); // 32 B contig
                V16 b; b.q[0] = wq[0]; b.q[1] = wq[1];
                acc = __builtin_amdgcn_wmma_f32_16x16x32_bf16(
                          false, afr[ks], false, b.v, (short)0, acc, false, false);
            }
            #pragma unroll
            for (int j = 0; j < 8; ++j) {
                const int row = j + half * 8;        // C/D: lanes 16-31 -> M+8
                sL[row * NOPT + o0 + m] = acc[j] * invT;
            }
        }
    }
    __syncthreads();

    // -------- softmax over 512 options; exps kept in registers --------------
    {
        const int tok = tid >> 4;                    // 16 lanes per token row
        const int sl  = tid & 15;                    // each owns 32 contiguous
        const float4* rp = (const float4*)(sL + tok * NOPT + sl * 32);

        float v[32];
        #pragma unroll
        for (int j = 0; j < 8; ++j) {                // 8x ds_load_b128
            const float4 t = rp[j];
            v[4 * j + 0] = t.x; v[4 * j + 1] = t.y;
            v[4 * j + 2] = t.z; v[4 * j + 3] = t.w;
        }
        float mx = v[0];
        #pragma unroll
        for (int i = 1; i < 32; ++i) mx = fmaxf(mx, v[i]);
        sR[tok * 16 + sl] = mx;
        __syncthreads();
        float rmax = sR[tok * 16];
        #pragma unroll
        for (int i = 1; i < 16; ++i) rmax = fmaxf(rmax, sR[tok * 16 + i]);

        float sum = 0.0f;
        #pragma unroll
        for (int i = 0; i < 32; ++i) {
            v[i] = __expf(v[i] - rmax);              // stays in registers
            sum += v[i];
        }
        __syncthreads();                             // all rmax reads done
        sR[tok * 16 + sl] = sum;
        __syncthreads();
        float rsum = 0.0f;
        #pragma unroll
        for (int i = 0; i < 16; ++i) rsum += sR[tok * 16 + i];
        const float rinv = 1.0f / rsum;

        uint4* pp = (uint4*)(sP + tok * NOPT + sl * 32);
        #pragma unroll
        for (int j = 0; j < 4; ++j) {                // 4x ds_store_b128
            PK8 pk;
            #pragma unroll
            for (int i = 0; i < 8; ++i) pk.h[i] = (bf16)(v[8 * j + i] * rinv);
            pp[j] = pk.q;
        }
    }
    __syncthreads();

    // -------- GEMM2: out[16 x 128] = probs[16 x 512] * C_h[512 x 128] -------
    {
        const int a0 = wave * 16;                    // 8 waves x 16 = 128 cols
        v8f acc = (v8f)0.0f;
        const bf16* crow = sC + (size_t)(a0 + m) * NOPT + half * 16;
        const bf16* prow = sP + m * NOPT + half * 16;
        #pragma unroll
        for (int ks = 0; ks < 16; ++ks) {
            const bf16* pa = prow + ks * 32;
            V16 a;                                   // 2x ds_load_b128
            a.q[0] = *(const uint4*)pa;
            a.q[1] = *(const uint4*)(pa + 8);
            const bf16* pb = crow + ks * 32;         // 32 B contiguous in LDS
            V16 b;
            b.q[0] = *(const uint4*)pb;
            b.q[1] = *(const uint4*)(pb + 8);
            acc = __builtin_amdgcn_wmma_f32_16x16x32_bf16(
                      false, a.v, false, b.v, (short)0, acc, false, false);
        }
        float* ob = out + ((size_t)t0 * NHEADS + h) * HS + a0;
        #pragma unroll
        for (int j = 0; j < 8; ++j) {
            const int row = j + half * 8;
            ob[(size_t)row * NHEADS * HS + m] = acc[j];
        }
    }
}

extern "C" void kernel_launch(void* const* d_in, const int* in_sizes, int n_in,
                              void* d_out, int out_size, void* d_ws, size_t ws_size,
                              hipStream_t stream) {
    (void)n_in; (void)out_size; (void)ws_size;     // requires ws_size >= 4 MB
    const float* x    = (const float*)d_in[0];
    const float* W    = (const float*)d_in[1];
    const float* C    = (const float*)d_in[2];
    const float* temp = (const float*)d_in[3];
    float* out = (float*)d_out;

    bf16* Wb = (bf16*)((char*)d_ws + WB_OFF);
    bf16* Ct = (bf16*)((char*)d_ws + CT_OFF);

    const int T = in_sizes[0] / (NHEADS * HS);     // B*S = 8192 tokens

    // Pass 1: weight/codebook conversion (1,048,576 float4 / 4 per thread)
    vq_prep<<<WC_ELEMS / 4 / 256, 256, 0, stream>>>(W, C, Wb, Ct);

    // Dynamic LDS > 64 KB needs the attribute bump (idempotent, host-side).
    hipFuncSetAttribute((const void*)vqizer_fused,
                        hipFuncAttributeMaxDynamicSharedMemorySize, SMEM_BYTES);

    dim3 grid(T / TT, NHEADS);
    vqizer_fused<<<grid, NTHREADS, SMEM_BYTES, stream>>>(x, Wb, Ct, temp, out);
}
